// BidGRU_60327110640348
// MI455X (gfx1250) — compile-verified, matrix operand order checked by reading, not run on hardware
//
#include <hip/hip_runtime.h>
#include <hip/hip_bf16.h>

typedef __attribute__((ext_vector_type(16))) __bf16 v16bf;
typedef __attribute__((ext_vector_type(8)))  __bf16 v8bf;
typedef __attribute__((ext_vector_type(8)))  float  v8f;
typedef __attribute__((ext_vector_type(4)))  float  v4f;
typedef __attribute__((ext_vector_type(4)))  int    v4i;

// gfx1250 async global->LDS path (guarded; falls back to sync path)
#if defined(__AMDGCN__) && __has_builtin(__builtin_amdgcn_global_load_async_to_lds_b128) && \
    __has_builtin(__builtin_amdgcn_s_wait_asynccnt)
#define USE_ASYNC_LDS 1
#endif

#ifdef USE_ASYNC_LDS
__device__ __forceinline__ void async_copy16(const __bf16* src, __bf16* dst) {
  // builtin signature (from hipcc diagnostic): (v4i*, v4i*, int, int)
  __builtin_amdgcn_global_load_async_to_lds_b128((v4i*)src, (v4i*)dst, 0, 0);
}
__device__ __forceinline__ void async_wait() { __builtin_amdgcn_s_wait_asynccnt(0); }
#else
__device__ __forceinline__ void async_copy16(const __bf16* src, __bf16* dst) {
  *(v8bf*)dst = *(const v8bf*)src;
}
__device__ __forceinline__ void async_wait() {}
#endif

// ---------------------------------------------------------------------------
// Convert f32 [K][N] -> bf16 [N][K] (transpose) so GEMM B rows are contiguous.
// ---------------------------------------------------------------------------
__global__ void transpose_to_bf16(const float* __restrict__ src,
                                  __bf16* __restrict__ dst, int K, int N) {
  size_t i = (size_t)blockIdx.x * blockDim.x + threadIdx.x;
  if (i >= (size_t)N * K) return;
  int n = (int)(i / K);
  int k = (int)(i % K);
  dst[i] = (__bf16)src[(size_t)k * N + n];
}

__global__ void zero_f32(float* __restrict__ p, size_t n) {
  size_t i = (size_t)blockIdx.x * blockDim.x + threadIdx.x;
  if (i < n) p[i] = 0.0f;
}

// ---------------------------------------------------------------------------
// Input-projection GEMM: C[M][N] = A_f32[M][K] * Bt_bf16[N][K]^T + bias
// BM=64, BN=128, BK=32; 256 threads = 8 waves; wave tile 16x64.
// ---------------------------------------------------------------------------
__global__ __launch_bounds__(256)
void gemm_bf16_kernel(const float* __restrict__ A, const __bf16* __restrict__ Bt,
                      const float* __restrict__ bias, float* __restrict__ C,
                      int M, int N, int K) {
  __shared__ __bf16 As[64][32];
  __shared__ __bf16 Bs[128][32];

  const int tid  = threadIdx.x;
  const int lane = tid & 31;
  const int wid  = tid >> 5;
  const int M0 = blockIdx.y * 64;
  const int N0 = blockIdx.x * 128;
  const int m0 = (wid & 3) * 16;
  const int n0 = (wid >> 2) * 64;
  const int g   = lane >> 4;
  const int l15 = lane & 15;

  v8f acc[4] = {};
  const int a_row = tid >> 2, a_seg = tid & 3;   // 64 rows x 4 chunks of 8
  const int b_row = tid >> 1, b_seg = tid & 1;   // 128 rows x 2 chunks of 16

  for (int kb = 0; kb < K; kb += 32) {
    // B tile: pure bf16 copy -> async global->LDS when available
    {
      const __bf16* bp = Bt + (size_t)(N0 + b_row) * K + kb + b_seg * 16;
      __bf16* dst = &Bs[b_row][b_seg * 16];
      async_copy16(bp, dst);
      async_copy16(bp + 8, dst + 8);
    }
    // A tile: f32 -> bf16 conversion through VGPRs
    {
      const float* ap = A + (size_t)(M0 + a_row) * K + kb + a_seg * 8;
      v4f a0 = *(const v4f*)ap;
      v4f a1 = *(const v4f*)(ap + 4);
      if (kb + 32 < K) __builtin_prefetch(ap + 32, 0, 1);
      __bf16* dst = &As[a_row][a_seg * 8];
      #pragma unroll
      for (int i = 0; i < 4; ++i) {
        dst[i]     = (__bf16)a0[i];
        dst[4 + i] = (__bf16)a1[i];
      }
    }
    async_wait();
    __syncthreads();

    v8bf alo = *(const v8bf*)&As[m0 + l15][g * 8];
    v8bf ahi = *(const v8bf*)&As[m0 + l15][16 + g * 8];
    v16bf av;
    #pragma unroll
    for (int i = 0; i < 8; ++i) { av[i] = alo[i]; av[8 + i] = ahi[i]; }

    v16bf bv[4];
    #pragma unroll
    for (int jn = 0; jn < 4; ++jn) {
      const __bf16* brow = &Bs[n0 + jn * 16 + l15][0];
      v8bf blo = *(const v8bf*)(brow + g * 8);
      v8bf bhi = *(const v8bf*)(brow + 16 + g * 8);
      #pragma unroll
      for (int i = 0; i < 8; ++i) { bv[jn][i] = blo[i]; bv[jn][8 + i] = bhi[i]; }
    }
    #pragma unroll
    for (int jn = 0; jn < 4; ++jn)
      acc[jn] = __builtin_amdgcn_wmma_f32_16x16x32_bf16(
          false, av, false, bv[jn], (short)0, acc[jn], false, false);
    __syncthreads();
  }

  #pragma unroll
  for (int jn = 0; jn < 4; ++jn) {
    const int col = N0 + n0 + jn * 16 + l15;
    const float bvs = bias[col];
    #pragma unroll
    for (int r = 0; r < 8; ++r) {
      const int row = M0 + m0 + r + g * 8;
      C[(size_t)row * N + col] = acc[jn][r] + bvs;
    }
  }
}

// ---------------------------------------------------------------------------
// Fully fused recurrent step (one launch per timestep, both directions).
// grid = (U/128, 2). Block owns u-slice [u0,u0+128); accumulates z/r/h~ gate
// GEMM tiles (hIn[64x512] @ Ut) via WMMA, then applies gates in registers:
//   z=sig(xz+hz), r=sig(xr+hr), c=tanh(xh+r*hh), h'=z*h+(1-z)*c (masked)
// Reads hIn, writes hOut (ping-pong buffers to avoid intra-launch races).
// ---------------------------------------------------------------------------
__global__ __launch_bounds__(256)
void gru_step_kernel(const float* __restrict__ xgF, const float* __restrict__ xgB,
                     const __bf16* __restrict__ UtF, const __bf16* __restrict__ UtB,
                     const float* __restrict__ b1F, const float* __restrict__ b1B,
                     const float* __restrict__ hInF, const float* __restrict__ hInB,
                     float* __restrict__ hOutF, float* __restrict__ hOutB,
                     const int* __restrict__ mask, float* __restrict__ out, int t) {
  constexpr int T = 250, Bsz = 64, U = 512, U3 = 1536, K = 512;
  __shared__ __bf16 As[64][32];        // 4 KB  (h tile, f32->bf16)
  __shared__ __bf16 Bs[3][128][32];    // 24 KB (z/r/h~ weight tiles)

  const int dir = blockIdx.y;
  const float*  xg   = dir ? xgB  : xgF;
  const __bf16* Ut   = dir ? UtB  : UtF;
  const float*  b1   = dir ? b1B  : b1F;
  const float*  hIn  = dir ? hInB : hInF;
  float*        hOut = dir ? hOutB : hOutF;
  const int te = dir ? (T - 1 - t) : t;
  const int u0 = blockIdx.x * 128;

  const int tid  = threadIdx.x;
  const int lane = tid & 31;
  const int wid  = tid >> 5;
  const int m0 = (wid & 3) * 16;
  const int n0 = (wid >> 2) * 64;
  const int g   = lane >> 4;
  const int l15 = lane & 15;

  v8f acc[3][4] = {};
  const int a_row = tid >> 2, a_seg = tid & 3;
  const int b_row = tid >> 1, b_seg = tid & 1;

  for (int kb = 0; kb < K; kb += 32) {
    #pragma unroll
    for (int gate = 0; gate < 3; ++gate) {
      const __bf16* bp = Ut + (size_t)(gate * U + u0 + b_row) * K + kb + b_seg * 16;
      __bf16* dst = &Bs[gate][b_row][b_seg * 16];
      async_copy16(bp, dst);
      async_copy16(bp + 8, dst + 8);
    }
    {
      const float* ap = hIn + (size_t)a_row * K + kb + a_seg * 8;
      v4f a0 = *(const v4f*)ap;
      v4f a1 = *(const v4f*)(ap + 4);
      __bf16* dst = &As[a_row][a_seg * 8];
      #pragma unroll
      for (int i = 0; i < 4; ++i) {
        dst[i]     = (__bf16)a0[i];
        dst[4 + i] = (__bf16)a1[i];
      }
    }
    async_wait();
    __syncthreads();

    v8bf alo = *(const v8bf*)&As[m0 + l15][g * 8];
    v8bf ahi = *(const v8bf*)&As[m0 + l15][16 + g * 8];
    v16bf av;
    #pragma unroll
    for (int i = 0; i < 8; ++i) { av[i] = alo[i]; av[8 + i] = ahi[i]; }

    #pragma unroll
    for (int gate = 0; gate < 3; ++gate) {
      v16bf bv[4];
      #pragma unroll
      for (int jn = 0; jn < 4; ++jn) {
        const __bf16* brow = &Bs[gate][n0 + jn * 16 + l15][0];
        v8bf blo = *(const v8bf*)(brow + g * 8);
        v8bf bhi = *(const v8bf*)(brow + 16 + g * 8);
        #pragma unroll
        for (int i = 0; i < 8; ++i) { bv[jn][i] = blo[i]; bv[jn][8 + i] = bhi[i]; }
      }
      #pragma unroll
      for (int jn = 0; jn < 4; ++jn)
        acc[gate][jn] = __builtin_amdgcn_wmma_f32_16x16x32_bf16(
            false, av, false, bv[jn], (short)0, acc[gate][jn], false, false);
    }
    __syncthreads();
  }

  // gate math + state update + outputs, all in registers
  #pragma unroll
  for (int jn = 0; jn < 4; ++jn) {
    const int u = u0 + n0 + jn * 16 + l15;              // 0..511
    const float bz = b1[u], br = b1[U + u], bh = b1[2 * U + u];
    #pragma unroll
    for (int r = 0; r < 8; ++r) {
      const int b = m0 + r + 8 * g;                     // batch row 0..63
      const size_t xrow = ((size_t)b * T + te) * U3;
      const float xz = xg[xrow + u];
      const float xr = xg[xrow + U + u];
      const float xh = xg[xrow + 2 * U + u];
      const float hz = acc[0][jn][r] + bz;
      const float hr = acc[1][jn][r] + br;
      const float hh = acc[2][jn][r] + bh;
      const float z  = 1.0f / (1.0f + __expf(-(xz + hz)));
      const float rg = 1.0f / (1.0f + __expf(-(xr + hr)));
      const float c  = tanhf(xh + rg * hh);
      const float ho = hIn[(size_t)b * U + u];
      float hn = z * ho + (1.0f - z) * c;
      hn = mask[b * T + te] ? hn : ho;
      hOut[(size_t)b * U + u] = hn;
      out[((size_t)b * T + te) * (2 * U) + dir * U + u] = hn;
      if (t == T - 1)
        out[(size_t)Bsz * T * 2 * U + (size_t)b * 2 * U + dir * U + u] = hn;
    }
  }
}

// ---------------------------------------------------------------------------
extern "C" void kernel_launch(void* const* d_in, const int* in_sizes, int n_in,
                              void* d_out, int out_size, void* d_ws, size_t ws_size,
                              hipStream_t stream) {
  constexpr int Bsz = 64, T = 250, C = 1024, U = 512, U3 = 1536;
  constexpr int M = Bsz * T;                      // 16000

  const float* x    = (const float*)d_in[0];
  const int*   mask = (const int*)d_in[1];
  const float* Wf   = (const float*)d_in[2];
  const float* Uf   = (const float*)d_in[3];
  const float* bf   = (const float*)d_in[4];      // [2][1536]
  const float* Wb   = (const float*)d_in[5];
  const float* Ub   = (const float*)d_in[6];
  const float* bb   = (const float*)d_in[7];
  float* out = (float*)d_out;

  // workspace layout
  float* xgF = (float*)d_ws;                      // 16000*1536
  float* xgB = xgF + (size_t)M * U3;
  float* hF0 = xgB + (size_t)M * U3;              // ping-pong h buffers
  float* hB0 = hF0 + (size_t)Bsz * U;
  float* hF1 = hB0 + (size_t)Bsz * U;
  float* hB1 = hF1 + (size_t)Bsz * U;
  __bf16* WtF = (__bf16*)(hB1 + (size_t)Bsz * U); // 1536*1024
  __bf16* WtB = WtF + (size_t)U3 * C;
  __bf16* UtF = WtB + (size_t)U3 * C;             // 1536*512
  __bf16* UtB = UtF + (size_t)U3 * U;

  // 1) weights -> bf16 [N][K]
  {
    size_t nW = (size_t)U3 * C, nU = (size_t)U3 * U;
    transpose_to_bf16<<<(nW + 255) / 256, 256, 0, stream>>>(Wf, WtF, C, U3);
    transpose_to_bf16<<<(nW + 255) / 256, 256, 0, stream>>>(Wb, WtB, C, U3);
    transpose_to_bf16<<<(nU + 255) / 256, 256, 0, stream>>>(Uf, UtF, U, U3);
    transpose_to_bf16<<<(nU + 255) / 256, 256, 0, stream>>>(Ub, UtB, U, U3);
  }
  // 2) zero parity-0 hidden states (hF0|hB0 contiguous)
  {
    size_t nH = (size_t)2 * Bsz * U;
    zero_f32<<<(nH + 255) / 256, 256, 0, stream>>>(hF0, nH);
  }
  // 3) input projections (M=16000, N=1536, K=1024)
  {
    dim3 grid(U3 / 128, M / 64);
    gemm_bf16_kernel<<<grid, 256, 0, stream>>>(x, WtF, bf, xgF, M, U3, C);
    gemm_bf16_kernel<<<grid, 256, 0, stream>>>(x, WtB, bb, xgB, M, U3, C);
  }
  // 4) recurrence: ONE fused kernel per step, ping-pong h buffers
  {
    dim3 grid(U / 128, 2);
    for (int t = 0; t < T; ++t) {
      const bool even = (t & 1) == 0;
      gru_step_kernel<<<grid, 256, 0, stream>>>(
          xgF, xgB, UtF, UtB, bf + U3, bb + U3,
          even ? hF0 : hF1, even ? hB0 : hB1,
          even ? hF1 : hF0, even ? hB1 : hB0,
          mask, out, t);
    }
  }
}
